// NLGTreeTrm_73400991088923
// MI455X (gfx1250) — matrix operand order repaired and checked by reading, not compiled, not run on hardware
//
#include <hip/hip_runtime.h>
#include <hip/hip_bf16.h>
#include <math.h>

// Problem constants (fixed by the reference).
#define L_NODES 128
#define B_SZ    512
#define D_SZ    512
#define H_SZ    512
#define KDIM    1536   // D + H + H (concat x | lh | rh)
#define NG      2560   // 5*H gate columns

// GEMM blocking: 128x128 block tile, 8 waves of 32x64, K-chunk 32, LDS double
// buffered. LDS row stride 40 elems (80B): 16B-aligned for b128 and bank-
// conflict-free (bank stride 20 across 16 fragment rows covers all 64 banks).
#define BM    128
#define BN    128
#define KC    32
#define LDST  40
#define LDS_TILE (BM * LDST)   // elems per buffer per operand

#if __has_builtin(__builtin_amdgcn_global_load_async_to_lds_b128)
#define HAS_ASYNC_LDS 1
#else
#define HAS_ASYNC_LDS 0
#endif

typedef __attribute__((ext_vector_type(16))) __bf16 v16bf;
typedef __attribute__((ext_vector_type(8)))  float  v8f;
typedef __attribute__((ext_vector_type(4)))  int    v4i;

typedef __attribute__((address_space(1))) v4i as1_v4i;   // global int4
typedef __attribute__((address_space(3))) v4i as3_v4i;   // LDS int4

union FragAB {
  v16bf v;
  uint4 q[2];
};

__device__ __forceinline__ unsigned short f2bf(float f) {
  // round-to-nearest-even f32 -> bf16
  unsigned int u = __builtin_bit_cast(unsigned int, f);
  u += 0x7FFFu + ((u >> 16) & 1u);
  return (unsigned short)(u >> 16);
}

__device__ __forceinline__ void async_wait0() {
#if __has_builtin(__builtin_amdgcn_s_wait_asynccnt)
  __builtin_amdgcn_s_wait_asynccnt(0);
#else
  asm volatile("s_wait_asynccnt 0" ::: "memory");
#endif
}

// ---------------------------------------------------------------------------
// Stage one 128x32 bf16 tile (global row-major, row stride K) into an LDS
// buffer with row stride LDST. 256 threads x 2 b128 chunks each.
// Async path: DMA straight into LDS (ASYNCcnt), no VGPR staging.
// ---------------------------------------------------------------------------
__device__ __forceinline__ void issue_tile_async(
    const unsigned short* __restrict__ G, int K, int row0, int k0,
    unsigned short* Sbuf)
{
#if HAS_ASYNC_LDS
#pragma unroll
  for (int i = 0; i < 2; ++i) {
    int idx = threadIdx.x * 2 + i;   // 0..511 ; 4 x b128 per 32-elem row
    int row = idx >> 2;
    int qc  = idx & 3;
    const unsigned short* g = G + (size_t)(row0 + row) * K + k0 + qc * 8;
    unsigned short*       l = Sbuf + row * LDST + qc * 8;
    __builtin_amdgcn_global_load_async_to_lds_b128(
        (as1_v4i*)g, (as3_v4i*)l, 0, 0);
  }
#else
  (void)G; (void)K; (void)row0; (void)k0; (void)Sbuf;
#endif
}

__device__ __forceinline__ void load_tile_regs(
    const unsigned short* __restrict__ G, int K, int row0, int k0, uint4 r[2])
{
#pragma unroll
  for (int i = 0; i < 2; ++i) {
    int idx = threadIdx.x * 2 + i;
    int row = idx >> 2;
    int qc  = idx & 3;
    const unsigned short* p = G + (size_t)(row0 + row) * K + k0 + qc * 8;
    r[i] = *(const uint4*)p;
    __builtin_prefetch(p + KC, 0, 3);
  }
}

__device__ __forceinline__ void store_tile_lds(unsigned short* S, uint4 r[2]) {
#pragma unroll
  for (int i = 0; i < 2; ++i) {
    int idx = threadIdx.x * 2 + i;
    int row = idx >> 2;
    int qc  = idx & 3;
    *(uint4*)(S + row * LDST + qc * 8) = r[i];
  }
}

// ---------------------------------------------------------------------------
// One K-chunk of compute: load ALL fragments (2xA, 4xB) from LDS first, then
// fire the 8 WMMAs as a single dependency-free burst (one dscnt wait, then
// back-to-back XDL issue).
// ---------------------------------------------------------------------------
__device__ __forceinline__ void compute_chunk(
    const unsigned short* __restrict__ Ab, const unsigned short* __restrict__ Bb,
    int mw, int nw, int lrow, bool hi, v8f acc[2][4])
{
  FragAB af[2], bf[4];
#pragma unroll
  for (int s = 0; s < 2; ++s) {
    const unsigned short* ar = Ab + (mw + s * 16 + lrow) * LDST + (hi ? 8 : 0);
    af[s].q[0] = *(const uint4*)(ar);        // K 0..7  (or 8..15)
    af[s].q[1] = *(const uint4*)(ar + 16);   // K 16..23 (or 24..31)
  }
#pragma unroll
  for (int j = 0; j < 4; ++j) {
    const unsigned short* br = Bb + (nw + j * 16 + lrow) * LDST + (hi ? 8 : 0);
    bf[j].q[0] = *(const uint4*)(br);        // K 0..7  (or 8..15)
    bf[j].q[1] = *(const uint4*)(br + 16);   // K 16..23 (or 24..31)
  }
#pragma unroll
  for (int j = 0; j < 4; ++j)
#pragma unroll
    for (int s = 0; s < 2; ++s)
      acc[s][j] = __builtin_amdgcn_wmma_f32_16x16x32_bf16(
          false, af[s].v, false, bf[j].v, (short)0, acc[s][j], false, false);
}

// ---------------------------------------------------------------------------
// Core: one 256-thread block computes a 128x128 f32 tile of A[MxK] @ W[NxK]^T
// in bf16 via v_wmma_f32_16x16x32_bf16.
// Async path: double-buffered GLOBAL_LOAD_ASYNC_TO_LDS_B128, one
// s_wait_asynccnt + one barrier per K-chunk (DMA overlaps the WMMA stream).
//   A fragment (16x32): lanes 0-15 row M=l K=0..7,16..23; lanes 16-31 K=8..15,24..31.
//   B fragment (32x16): lane n = W row n; lanes 0-15 K=0..7,16..23; lanes 16-31 K=8..15,24..31.
// ---------------------------------------------------------------------------
__device__ __forceinline__ void gemm_core_128x128(
    const unsigned short* __restrict__ A,   // [M x K] row-major bf16 bits
    const unsigned short* __restrict__ W,   // [N x K] row-major bf16 bits
    int K, int am0, int wn0,
    unsigned short* As, unsigned short* Bs,  // [2][LDS_TILE] each
    v8f acc[2][4])
{
  const int  lane = threadIdx.x & 31;
  const int  w    = threadIdx.x >> 5;
  const int  mw   = (w & 3) * 32;
  const int  nw   = (w >> 2) * 64;
  const int  lrow = lane & 15;
  const bool hi   = lane >= 16;

#if HAS_ASYNC_LDS
  issue_tile_async(A, K, am0, 0, As);
  issue_tile_async(W, K, wn0, 0, Bs);
  async_wait0();
  __syncthreads();

  int buf = 0;
  for (int k0 = 0; k0 < K; k0 += KC) {
    const bool has_next = (k0 + KC) < K;
    if (has_next) {
      // DMA next chunk into the other buffer while we compute this one.
      issue_tile_async(A, K, am0, k0 + KC, As + (1 - buf) * LDS_TILE);
      issue_tile_async(W, K, wn0, k0 + KC, Bs + (1 - buf) * LDS_TILE);
    }
    compute_chunk(As + buf * LDS_TILE, Bs + buf * LDS_TILE,
                  mw, nw, lrow, hi, acc);
    if (has_next) {
      async_wait0();      // my DMA writes into (1-buf) have landed
      __syncthreads();    // everyone's landed; everyone done reading (buf)
      buf ^= 1;
    }
  }
#else
  uint4 ra[2], rb[2];
  load_tile_regs(A, K, am0, 0, ra);
  load_tile_regs(W, K, wn0, 0, rb);
  store_tile_lds(As, ra);
  store_tile_lds(Bs, rb);
  __syncthreads();

  int buf = 0;
  for (int k0 = 0; k0 < K; k0 += KC) {
    const bool has_next = (k0 + KC) < K;
    if (has_next) {
      load_tile_regs(A, K, am0, k0 + KC, ra);
      load_tile_regs(W, K, wn0, k0 + KC, rb);
    }
    compute_chunk(As + buf * LDS_TILE, Bs + buf * LDS_TILE,
                  mw, nw, lrow, hi, acc);
    if (has_next) {
      __syncthreads();
      store_tile_lds(As + (1 - buf) * LDS_TILE, ra);
      store_tile_lds(Bs + (1 - buf) * LDS_TILE, rb);
      __syncthreads();
      buf ^= 1;
    }
  }
#endif
}

// ---------------------------------------------------------------------------
// Per-step kernel 1: gather lh/rh (child averages), build bf16 Xg = [x|lh|rh]
// and the x-slab of Xc; bump the parent's child counters (consumed only by
// LATER steps, so no ordering hazard inside this step).
// ---------------------------------------------------------------------------
__global__ void prep_kernel(
    const float* __restrict__ inputs,   // [L x B x D]
    const int*   __restrict__ idx_row,  // [B] node id per batch, this step
    const int*   __restrict__ parent,   // [B x L]
    const int*   __restrict__ is_left,  // [B x L]
    const float* __restrict__ lsum,     // [B x L x H]
    const float* __restrict__ rsum,
    int*   __restrict__ lcnt,           // [B x L]
    int*   __restrict__ rcnt,
    float* __restrict__ lh,             // [B x H]
    float* __restrict__ rh,
    unsigned short* __restrict__ Xg,    // [B x KDIM]
    unsigned short* __restrict__ Xc)    // [B x KDIM]
{
  const int b      = blockIdx.x;
  const int si     = idx_row[b];
  const bool valid = si >= 0;
  const int s      = valid ? si : 0;

  const int lc = lcnt[b * L_NODES + s];
  const int rc = rcnt[b * L_NODES + s];
  const float linv = (valid && lc > 0) ? 1.0f / (float)lc : 0.0f;
  const float rinv = (valid && rc > 0) ? 1.0f / (float)rc : 0.0f;
  const size_t nb = (size_t)b * L_NODES + s;

  for (int c = threadIdx.x; c < H_SZ; c += blockDim.x) {
    float x = inputs[((size_t)s * B_SZ + b) * D_SZ + c];
    unsigned short xb = f2bf(x);
    Xg[(size_t)b * KDIM + c] = xb;
    Xc[(size_t)b * KDIM + c] = xb;
    float lv = lsum[nb * H_SZ + c] * linv;
    float rv = rsum[nb * H_SZ + c] * rinv;
    lh[(size_t)b * H_SZ + c] = lv;
    rh[(size_t)b * H_SZ + c] = rv;
    Xg[(size_t)b * KDIM +     H_SZ + c] = f2bf(lv);
    Xg[(size_t)b * KDIM + 2 * H_SZ + c] = f2bf(rv);
  }

  if (threadIdx.x == 0 && valid) {
    int p = parent[b * L_NODES + si];
    if (p >= 0) {
      if (is_left[b * L_NODES + si]) lcnt[b * L_NODES + p] += 1;
      else                           rcnt[b * L_NODES + p] += 1;
    }
  }
}

// ---------------------------------------------------------------------------
// Per-step kernel 2: gates = sigmoid(Xg @ Wg^T + b_gih)   [512 x 2560]
// Epilogue also builds Xc[:,512:1536] = [lh*rl | rh*rr] in bf16.
// Grid: (512/128)*(2560/128) = 4*20 = 80 blocks x 256 threads.
// ---------------------------------------------------------------------------
__global__ void gemm_gates_kernel(
    const unsigned short* __restrict__ Xg,   // [B x KDIM]
    const unsigned short* __restrict__ Wg,   // [NG x KDIM]
    const float* __restrict__ b_gih,         // [NG]
    const float* __restrict__ lh,            // [B x H]
    const float* __restrict__ rh,
    float* __restrict__ gates,               // [B x NG]
    unsigned short* __restrict__ Xc)         // [B x KDIM]
{
  __shared__ unsigned short As[2 * LDS_TILE];
  __shared__ unsigned short Bs[2 * LDS_TILE];

  const int bm = blockIdx.x / (NG / BN);   // 0..3
  const int bn = blockIdx.x % (NG / BN);   // 0..19
  const int am0 = bm * BM;
  const int wn0 = bn * BN;

  v8f acc[2][4] = {};
  gemm_core_128x128(Xg, Wg, KDIM, am0, wn0, As, Bs, acc);

  const int lane = threadIdx.x & 31;
  const int w    = threadIdx.x >> 5;
  const int mw   = (w & 3) * 32;
  const int nw   = (w >> 2) * 64;
  const int lrow = lane & 15;
  const int hofs = (lane >= 16) ? 8 : 0;

#pragma unroll
  for (int s = 0; s < 2; ++s) {
    const int mbase = am0 + mw + s * 16 + hofs;
#pragma unroll
    for (int j = 0; j < 4; ++j) {
      const int n = wn0 + nw + j * 16 + lrow;
      const float bias = b_gih[n];
#pragma unroll
      for (int r = 0; r < 8; ++r) {
        const int m = mbase + r;
        float g = 1.0f / (1.0f + __expf(-(acc[s][j][r] + bias)));
        gates[(size_t)m * NG + n] = g;
        if (n < H_SZ) {               // rl region -> Xc[:, 512+n] = lh*rl
          Xc[(size_t)m * KDIM + H_SZ + n] = f2bf(lh[(size_t)m * H_SZ + n] * g);
        } else if (n < 2 * H_SZ) {    // rr region -> Xc[:, 1024+n'] = rh*rr
          const int n2 = n - H_SZ;
          Xc[(size_t)m * KDIM + 2 * H_SZ + n2] =
              f2bf(rh[(size_t)m * H_SZ + n2] * g);
        }
      }
    }
  }
}

// ---------------------------------------------------------------------------
// Per-step kernel 3: cell = tanh(Xc @ Wc^T + b_cih); h = zl*lh + zr*rh + z*cell
// Write h_all, scatter-add h into parent's lsum/rsum (row-private per batch,
// so plain adds are race-free within a step).
// Grid: (512/128)*(512/128) = 16 blocks x 256 threads.
// ---------------------------------------------------------------------------
__global__ void gemm_cell_kernel(
    const unsigned short* __restrict__ Xc,   // [B x KDIM]
    const unsigned short* __restrict__ Wc,   // [H x KDIM]
    const float* __restrict__ b_cih,         // [H]
    const float* __restrict__ gates,         // [B x NG]
    const float* __restrict__ lh,            // [B x H]
    const float* __restrict__ rh,
    const int*   __restrict__ idx_row,       // [B]
    const int*   __restrict__ parent,        // [B x L]
    const int*   __restrict__ is_left,       // [B x L]
    float* __restrict__ h_all,               // [B x L x H] (in d_out)
    float* __restrict__ lsum,                // [B x L x H]
    float* __restrict__ rsum)
{
  __shared__ unsigned short As[2 * LDS_TILE];
  __shared__ unsigned short Bs[2 * LDS_TILE];

  const int bm = blockIdx.x / (H_SZ / BN);   // 0..3
  const int bn = blockIdx.x % (H_SZ / BN);   // 0..3
  const int am0 = bm * BM;
  const int wn0 = bn * BN;

  v8f acc[2][4] = {};
  gemm_core_128x128(Xc, Wc, KDIM, am0, wn0, As, Bs, acc);

  const int lane = threadIdx.x & 31;
  const int w    = threadIdx.x >> 5;
  const int mw   = (w & 3) * 32;
  const int nw   = (w >> 2) * 64;
  const int lrow = lane & 15;
  const int hofs = (lane >= 16) ? 8 : 0;

#pragma unroll
  for (int s = 0; s < 2; ++s) {
    const int mbase = am0 + mw + s * 16 + hofs;
#pragma unroll
    for (int j = 0; j < 4; ++j) {
      const int n = wn0 + nw + j * 16 + lrow;
      const float bias = b_cih[n];
#pragma unroll
      for (int r = 0; r < 8; ++r) {
        const int m = mbase + r;
        const int si = idx_row[m];
        if (si < 0) continue;
        float cell = tanhf(acc[s][j][r] + bias);
        const size_t gb = (size_t)m * NG;
        const float lhv = lh[(size_t)m * H_SZ + n];
        const float rhv = rh[(size_t)m * H_SZ + n];
        float h = gates[gb + 2 * H_SZ + n] * lhv   // zl
                + gates[gb + 3 * H_SZ + n] * rhv   // zr
                + gates[gb + 4 * H_SZ + n] * cell; // z
        h_all[((size_t)m * L_NODES + si) * H_SZ + n] = h;
        const int p = parent[m * L_NODES + si];
        if (p >= 0) {
          const size_t off = ((size_t)m * L_NODES + p) * H_SZ + n;
          if (is_left[m * L_NODES + si]) lsum[off] += h;
          else                           rsum[off] += h;
        }
      }
    }
  }
}

// ---------------------------------------------------------------------------
// One-time per launch: zero accumulators / output; pack weights to bf16 as
// Wg = [W_gih | W_glh | W_grh]  (2560 x 1536)  and
// Wc = [W_cih | W_clh | W_crh]  ( 512 x 1536), both row-major over K.
// ---------------------------------------------------------------------------
__global__ void zero_u32_kernel(unsigned int* __restrict__ p, size_t n) {
  size_t i = (size_t)blockIdx.x * blockDim.x + threadIdx.x;
  size_t stride = (size_t)gridDim.x * blockDim.x;
  for (; i < n; i += stride) p[i] = 0u;
}

__global__ void convert_weights_kernel(
    const float* __restrict__ Wgih, const float* __restrict__ Wglh,
    const float* __restrict__ Wgrh, const float* __restrict__ Wcih,
    const float* __restrict__ Wclh, const float* __restrict__ Wcrh,
    unsigned short* __restrict__ Wg, unsigned short* __restrict__ Wc)
{
  const size_t totalG = (size_t)NG * KDIM;
  const size_t totalC = (size_t)H_SZ * KDIM;
  size_t i = (size_t)blockIdx.x * blockDim.x + threadIdx.x;
  size_t stride = (size_t)gridDim.x * blockDim.x;
  for (size_t t = i; t < totalG + totalC; t += stride) {
    if (t < totalG) {
      size_t nr = t / KDIM, c = t % KDIM;
      float v = (c < H_SZ)     ? Wgih[nr * D_SZ + c]
              : (c < 2 * H_SZ) ? Wglh[nr * H_SZ + (c - H_SZ)]
                               : Wgrh[nr * H_SZ + (c - 2 * H_SZ)];
      Wg[t] = f2bf(v);
    } else {
      size_t t2 = t - totalG;
      size_t nr = t2 / KDIM, c = t2 % KDIM;
      float v = (c < D_SZ)     ? Wcih[nr * D_SZ + c]
              : (c < 2 * H_SZ) ? Wclh[nr * H_SZ + (c - H_SZ)]
                               : Wcrh[nr * H_SZ + (c - 2 * H_SZ)];
      Wc[t2] = f2bf(v);
    }
  }
}

// ---------------------------------------------------------------------------
extern "C" void kernel_launch(void* const* d_in, const int* in_sizes, int n_in,
                              void* d_out, int out_size, void* d_ws, size_t ws_size,
                              hipStream_t stream) {
  (void)in_sizes; (void)n_in; (void)ws_size;

  const float* inputs  = (const float*)d_in[0];   // [L,B,D]
  const int*   indexes = (const int*)  d_in[1];   // [L,B]
  const int*   parent  = (const int*)  d_in[2];   // [B,L]
  const int*   is_left = (const int*)  d_in[3];   // [B,L]
  const float* W_gih   = (const float*)d_in[4];
  const float* b_gih   = (const float*)d_in[5];
  const float* W_glh   = (const float*)d_in[6];
  const float* W_grh   = (const float*)d_in[7];
  const float* W_cih   = (const float*)d_in[8];
  const float* b_cih   = (const float*)d_in[9];
  const float* W_clh   = (const float*)d_in[10];
  const float* W_crh   = (const float*)d_in[11];

  float* out = (float*)d_out;   // h_all [B,L,H] then output_t [B,H] (zeros)

  // Workspace carve-up (256B aligned).
  char* ws = (char*)d_ws;
  size_t off = 0;
  auto carve = [&](size_t bytes) {
    void* p = ws + off;
    off = (off + bytes + 255) & ~(size_t)255;
    return p;
  };
  const size_t BLH = (size_t)B_SZ * L_NODES * H_SZ;
  float*          lsum  = (float*)carve(BLH * 4);
  float*          rsum  = (float*)carve(BLH * 4);
  int*            lcnt  = (int*)  carve((size_t)B_SZ * L_NODES * 4);
  int*            rcnt  = (int*)  carve((size_t)B_SZ * L_NODES * 4);
  float*          lh    = (float*)carve((size_t)B_SZ * H_SZ * 4);
  float*          rh    = (float*)carve((size_t)B_SZ * H_SZ * 4);
  float*          gates = (float*)carve((size_t)B_SZ * NG * 4);
  unsigned short* Xg    = (unsigned short*)carve((size_t)B_SZ * KDIM * 2);
  unsigned short* Xc    = (unsigned short*)carve((size_t)B_SZ * KDIM * 2);
  unsigned short* Wg    = (unsigned short*)carve((size_t)NG * KDIM * 2);
  unsigned short* Wc    = (unsigned short*)carve((size_t)H_SZ * KDIM * 2);

  // Zero the recurrent accumulators + counters (contiguous carve) and d_out.
  const size_t zero_words = 2 * BLH + 2 * (size_t)B_SZ * L_NODES;
  zero_u32_kernel<<<4096, 256, 0, stream>>>((unsigned int*)lsum, zero_words);
  zero_u32_kernel<<<4096, 256, 0, stream>>>((unsigned int*)out, (size_t)out_size);
  convert_weights_kernel<<<2048, 256, 0, stream>>>(W_gih, W_glh, W_grh,
                                                   W_cih, W_clh, W_crh, Wg, Wc);

  // G1: 4x20 = 80 block tiles of 128x128. G2: 4x4 = 16 block tiles.
  for (int t = 0; t < L_NODES; ++t) {
    const int* idx_row = indexes + (size_t)t * B_SZ;
    prep_kernel<<<B_SZ, 256, 0, stream>>>(inputs, idx_row, parent, is_left,
                                          lsum, rsum, lcnt, rcnt, lh, rh, Xg, Xc);
    gemm_gates_kernel<<<(B_SZ / BM) * (NG / BN), 256, 0, stream>>>(
        Xg, Wg, b_gih, lh, rh, gates, Xc);
    gemm_cell_kernel<<<(B_SZ / BM) * (H_SZ / BN), 256, 0, stream>>>(
        Xc, Wc, b_cih, gates, lh, rh, idx_row, parent, is_left,
        out /*h_all*/, lsum, rsum);
  }
}
